// LatentAttnProcessor_56908316672229
// MI455X (gfx1250) — matrix operand order repaired and checked by reading, not compiled, not run on hardware
//
#include <hip/hip_runtime.h>

typedef _Float16 h16;
typedef __attribute__((ext_vector_type(16))) _Float16 v16h;
typedef __attribute__((ext_vector_type(8)))  float    v8f;
typedef __attribute__((ext_vector_type(4)))  unsigned int v4u;
typedef __attribute__((ext_vector_type(8)))  int v8i_;
typedef __attribute__((ext_vector_type(4)))  int v4i_;

#define HEADS 8
#define DHEAD 64
#define NTOK  4096
#define QD    1280
#define INNER 512
#define ATT_SCALE 0.125f   // 64^-0.5

#if defined(__AMDGCN__) && __has_builtin(__builtin_amdgcn_tensor_load_to_lds) && __has_builtin(__builtin_amdgcn_s_wait_tensorcnt)
#define HAVE_TDM 1
#else
#define HAVE_TDM 0
#endif

// 16-bit A-matrix (16x32) K offset for half-index h, lane-half hi
__device__ __forceinline__ int a_koff(int h, int hi) {
    return ((h < 8) ? h : h + 8) + (hi ? 8 : 0);
}

__device__ __forceinline__ v8f wmma_f16(v16h a, v16h b, v8f c) {
    return __builtin_amdgcn_wmma_f32_16x16x32_f16(false, a, false, b, (short)0, c, false, false);
}

// ---------------------------------------------------------------------------
// Pack activations fp32 -> f16 (one-time conversion; kills per-tile cvt spam)
// ---------------------------------------------------------------------------
__global__ __launch_bounds__(256) void pack_x_kernel(const float* __restrict__ hs,
                                                     const float* __restrict__ ehs,
                                                     h16* __restrict__ Xh,
                                                     h16* __restrict__ Eh) {
    size_t idx = (size_t)blockIdx.x * blockDim.x + threadIdx.x;
    if (idx >= (size_t)NTOK * QD) return;
    const float* src = blockIdx.y ? ehs : hs;
    h16*         dst = blockIdx.y ? Eh  : Xh;
    dst[idx] = (h16)src[idx];
}

// ---------------------------------------------------------------------------
// Pack weights fp32 -> f16 in B-fragment-native layout:
// out[((kt*(N/16)+nt)*32 + lane)*16 + h] = W[(kt*32 + (h<8?h:h+8) + 8*(lane>=16))*N + nt*16 + (lane&15)]
// => in GEMM each lane does one aligned 32B v16h load, coalesced across lanes.
// ---------------------------------------------------------------------------
__global__ __launch_bounds__(128) void pack_w_kernel(const float* __restrict__ W,
                                                     h16* __restrict__ out,
                                                     int Kdim, int N) {
    int gt = blockIdx.x * blockDim.x + threadIdx.x;
    int lane = gt & 31;
    int tile = gt >> 5;
    int nt16 = N >> 4;
    int numTiles = (Kdim >> 5) * nt16;
    if (tile >= numTiles) return;
    int kt = tile / nt16, nt = tile % nt16;
    int hi = (lane >> 4) & 1, ln = lane & 15;
    h16* dst = out + ((size_t)tile * 32 + lane) * 16;
    const float* src = W + (size_t)(kt * 32 + 8 * hi) * N + nt * 16 + ln;
    #pragma unroll
    for (int h = 0; h < 16; ++h) dst[h] = (h16)src[(size_t)(h < 8 ? h : h + 8) * N];
}

// ---------------------------------------------------------------------------
// Mask: bilinear 512x512 -> 64x64 (align_corners=False), binarize 1->5, 0->0.1
// ---------------------------------------------------------------------------
__global__ __launch_bounds__(256) void mask_kernel(const float* __restrict__ lam,
                                                   float* __restrict__ maskout) {
    int idx = blockIdx.x * blockDim.x + threadIdx.x;
    if (idx >= NTOK) return;
    int i = idx >> 6, j = idx & 63;
    int y = 8 * i + 3, x = 8 * j + 3;
    float v = 0.25f * (lam[y * 512 + x] + lam[y * 512 + x + 1] +
                       lam[(y + 1) * 512 + x] + lam[(y + 1) * 512 + x + 1]);
    if (v == 1.0f) v = 5.0f;
    else if (v == 0.0f) v = 0.1f;
    maskout[idx] = v;
}

// ---------------------------------------------------------------------------
// QKV projection: Xf16(4096x1280) @ Wpacked -> Q,K f16 row-major; V stored
// transposed (Vt[dcol][token]) so attention P@V B-fragments are contiguous.
// ---------------------------------------------------------------------------
__global__ __launch_bounds__(128) void gemm_qkv_kernel(
    const h16* __restrict__ Xh, const h16* __restrict__ Eh,
    const h16* __restrict__ Wqp, const h16* __restrict__ Wkp,
    const h16* __restrict__ Wvp,
    h16* __restrict__ Qb, h16* __restrict__ Kb, h16* __restrict__ Vt)
{
    const int lane = threadIdx.x & 31;
    const int wave = threadIdx.x >> 5;
    const int hi   = (lane >> 4) & 1;
    const int ln   = lane & 15;

    const h16* X; const h16* Wp; h16* Out; bool vtrans = false;
    if (blockIdx.z == 0)      { X = Xh; Wp = Wqp; Out = Qb; }
    else if (blockIdx.z == 1) { X = Eh; Wp = Wkp; Out = Kb; }
    else                      { X = Eh; Wp = Wvp; Out = Vt; vtrans = true; }

    const int m0 = blockIdx.x * 16;
    const int n0 = blockIdx.y * 256 + wave * 64;
    const int NT16 = INNER >> 4;

    v8f acc[4] = {};
    const h16* xrow = X + (size_t)(m0 + ln) * QD;

    for (int kb = 0; kb < QD; kb += 32) {
        v16h a;
        const h16* xp = xrow + kb + 8 * hi;
        #pragma unroll
        for (int h = 0; h < 8; ++h) { a[h] = xp[h]; a[h + 8] = xp[h + 16]; }
        #pragma unroll
        for (int t = 0; t < 4; ++t) {
            v16h b = *(const v16h*)(Wp + (((size_t)(kb >> 5) * NT16 + (n0 >> 4) + t) * 32 + lane) * 16);
            acc[t] = wmma_f16(a, b, acc[t]);
        }
    }
    #pragma unroll
    for (int t = 0; t < 4; ++t)
        #pragma unroll
        for (int r = 0; r < 8; ++r) {
            h16 val = (h16)acc[t][r];
            if (vtrans) Out[(size_t)(n0 + t * 16 + ln) * NTOK + (m0 + r + 8 * hi)] = val;
            else        Out[(size_t)(m0 + r + 8 * hi) * INNER + (n0 + t * 16 + ln)] = val;
        }
}

// ---------------------------------------------------------------------------
// Flash attention. Block = 4 waves, one head, 4 adjacent 16-row q blocks.
// Shared 32x64 K tile staged in LDS via TDM (tensor_load_to_lds) when available.
// ---------------------------------------------------------------------------
__global__ __launch_bounds__(128) void attn_kernel(
    const h16* __restrict__ Qb, const h16* __restrict__ Kb, const h16* __restrict__ Vt,
    const float* __restrict__ maskp, h16* __restrict__ Oh)
{
    __shared__ h16 shK[32 * 64];      // 4 KB K tile (rows j, cols d)
    __shared__ h16 psh[4][16 * 32];   // per-wave P transpose staging (4 KB)

    const int lane = threadIdx.x & 31;
    const int wave = threadIdx.x >> 5;
    const int hi   = (lane >> 4) & 1;
    const int ln   = lane & 15;

    const int head = blockIdx.x >> 6;
    const int m0   = (blockIdx.x & 63) * 64 + wave * 16;

    // Q as two 16x32 A fragments (d = 0..31, 32..63): contiguous 16B runs
    v16h aq0, aq1;
    const h16* qrow = Qb + (size_t)(m0 + ln) * INNER + head * DHEAD + 8 * hi;
    #pragma unroll
    for (int h = 0; h < 8; ++h) {
        aq0[h] = qrow[h];      aq0[h + 8] = qrow[h + 16];
        aq1[h] = qrow[h + 32]; aq1[h + 8] = qrow[h + 48];
    }

    float rm[8], rl[8], smask[8];
    #pragma unroll
    for (int r = 0; r < 8; ++r) {
        rm[r] = -1e30f;
        rl[r] = 0.0f;
        smask[r] = maskp[m0 + r + 8 * hi] * ATT_SCALE;
    }
    v8f o[4] = {};

    for (int j0 = 0; j0 < NTOK; j0 += 32) {
        __syncthreads();   // previous tile fully consumed
#if HAVE_TDM
        if (threadIdx.x < 32) {   // wave 0 issues the TDM load (EXEC ignored by TDM)
            unsigned long long ga =
                (unsigned long long)(uintptr_t)(Kb + (size_t)j0 * INNER + head * DHEAD);
            unsigned int lds = (unsigned int)(uintptr_t)&shK[0];
            v4u g0;
            g0.x = 1u;                                              // count=1
            g0.y = lds;                                             // lds_addr
            g0.z = (unsigned int)(ga & 0xFFFFFFFFu);                // global_addr[31:0]
            g0.w = (unsigned int)((ga >> 32) & 0x1FFFFFFu) | (2u << 30); // ga[56:32] | type=2
            v8i_ g1;
            g1[0] = (int)(1u << 16);      // workgroup_mask=0, data_size=2B
            g1[1] = (int)(64u << 16);     // tensor_dim0 = 64
            g1[2] = (int)(4096u << 16);   // tensor_dim1 = 4096
            g1[3] = (int)(64u << 16);     // tile_dim0 = 64
            g1[4] = 32;                   // tile_dim1 = 32, tile_dim2 = 0
            g1[5] = (int)INNER;           // tensor_dim0_stride = 512
            g1[6] = 0;
            g1[7] = 0;
            v4i_ gz4 = {0, 0, 0, 0};
            v8i_ gz8 = {0, 0, 0, 0, 0, 0, 0, 0};
            __builtin_amdgcn_tensor_load_to_lds(g0, g1, gz4, gz4, gz8, 0);
            __builtin_amdgcn_s_wait_tensorcnt(0);
        }
#else
        {   // cooperative fallback copy: 128 threads x 16 halves = 32x64 tile
            int t   = threadIdx.x;
            int row = t >> 2, seg = (t & 3) * 16;
            const h16* src = Kb + (size_t)(j0 + row) * INNER + head * DHEAD + seg;
            h16* dst = &shK[row * 64 + seg];
            #pragma unroll
            for (int h = 0; h < 16; ++h) dst[h] = src[h];
        }
#endif
        __syncthreads();   // tile visible to all 4 waves

        // S = Q @ K^T : two 16x16 tiles (4 WMMAs), K B-fragments from LDS
        v8f st[2];
        #pragma unroll
        for (int s = 0; s < 2; ++s) {
            const h16* kp = &shK[(16 * s + ln) * 64 + 8 * hi];
            v16h b0, b1;
            #pragma unroll
            for (int h = 0; h < 16; ++h) { b0[h] = kp[a_koff(h, 0)]; b1[h] = kp[a_koff(h, 0) + 32]; }
            v8f z = {};
            z = wmma_f16(aq0, b0, z);
            z = wmma_f16(aq1, b1, z);
            st[s] = z;
        }

        // online softmax update (rows live within 16-lane halves)
        #pragma unroll
        for (int r = 0; r < 8; ++r) {
            float v0 = st[0][r] * smask[r];
            float v1 = st[1][r] * smask[r];
            float mx = fmaxf(v0, v1);
            #pragma unroll
            for (int d = 1; d < 16; d <<= 1) mx = fmaxf(mx, __shfl_xor(mx, d, 32));
            float nm = fmaxf(rm[r], mx);
            float cf = __expf(rm[r] - nm);
            v0 = __expf(v0 - nm);
            v1 = __expf(v1 - nm);
            float sum = v0 + v1;
            #pragma unroll
            for (int d = 1; d < 16; d <<= 1) sum += __shfl_xor(sum, d, 32);
            rl[r] = rl[r] * cf + sum;
            rm[r] = nm;
            o[0][r] *= cf; o[1][r] *= cf; o[2][r] *= cf; o[3][r] *= cf;
            st[0][r] = v0; st[1][r] = v1;
        }

        // P (C layout) -> LDS -> A-fragment layout
        #pragma unroll
        for (int s = 0; s < 2; ++s)
            #pragma unroll
            for (int r = 0; r < 8; ++r)
                psh[wave][(r + 8 * hi) * 32 + 16 * s + ln] = (h16)st[s][r];
        v16h ap;
        #pragma unroll
        for (int h = 0; h < 16; ++h) ap[h] = psh[wave][ln * 32 + 16 * hi + h];

        // O += P @ V : B-fragments contiguous from Vt (d-major)
        #pragma unroll
        for (int t = 0; t < 4; ++t) {
            v16h bv;
            const h16* vp = Vt + (size_t)(head * DHEAD + t * 16 + ln) * NTOK + j0 + 16 * hi;
            #pragma unroll
            for (int h = 0; h < 16; ++h) bv[h] = vp[h];
            o[t] = wmma_f16(ap, bv, o[t]);
        }
    }

    #pragma unroll
    for (int r = 0; r < 8; ++r) rl[r] = 1.0f / rl[r];
    #pragma unroll
    for (int t = 0; t < 4; ++t)
        #pragma unroll
        for (int r = 0; r < 8; ++r)
            Oh[(size_t)(m0 + r + 8 * hi) * INNER + head * DHEAD + t * 16 + ln] =
                (h16)(o[t][r] * rl[r]);
}

// ---------------------------------------------------------------------------
// Output projection: O_f16(4096x512) @ Wo_packed(512x1280) + bo -> fp32 out
// ---------------------------------------------------------------------------
__global__ __launch_bounds__(128) void gemm_out_kernel(
    const h16* __restrict__ Oh, const h16* __restrict__ Wop,
    const float* __restrict__ bias, float* __restrict__ Out)
{
    const int lane = threadIdx.x & 31;
    const int wave = threadIdx.x >> 5;
    const int hi   = (lane >> 4) & 1;
    const int ln   = lane & 15;

    const int m0 = blockIdx.x * 16;
    const int n0 = blockIdx.y * 256 + wave * 64;
    const int NT16 = QD >> 4;

    v8f acc[4] = {};
    const h16* xrow = Oh + (size_t)(m0 + ln) * INNER;

    for (int kb = 0; kb < INNER; kb += 32) {
        v16h a;
        const h16* xp = xrow + kb + 8 * hi;
        #pragma unroll
        for (int h = 0; h < 8; ++h) { a[h] = xp[h]; a[h + 8] = xp[h + 16]; }
        #pragma unroll
        for (int t = 0; t < 4; ++t) {
            v16h b = *(const v16h*)(Wop + (((size_t)(kb >> 5) * NT16 + (n0 >> 4) + t) * 32 + lane) * 16);
            acc[t] = wmma_f16(a, b, acc[t]);
        }
    }
    #pragma unroll
    for (int t = 0; t < 4; ++t) {
        float bn = bias[n0 + t * 16 + ln];
        #pragma unroll
        for (int r = 0; r < 8; ++r)
            Out[(size_t)(m0 + r + 8 * hi) * QD + (n0 + t * 16 + ln)] = acc[t][r] + bn;
    }
}

// ---------------------------------------------------------------------------
extern "C" void kernel_launch(void* const* d_in, const int* in_sizes, int n_in,
                              void* d_out, int out_size, void* d_ws, size_t ws_size,
                              hipStream_t stream) {
    (void)in_sizes; (void)n_in; (void)out_size; (void)ws_size;
    const float* hs  = (const float*)d_in[0];
    const float* ehs = (const float*)d_in[1];
    const float* lam = (const float*)d_in[3];
    const float* Wq  = (const float*)d_in[4];
    const float* Wk  = (const float*)d_in[5];
    const float* Wv  = (const float*)d_in[6];
    const float* Wo  = (const float*)d_in[7];
    const float* bo  = (const float*)d_in[8];
    float* out = (float*)d_out;

    h16* Xh  = (h16*)d_ws;
    h16* Eh  = Xh  + (size_t)NTOK * QD;
    h16* Qb  = Eh  + (size_t)NTOK * QD;
    h16* Kb  = Qb  + (size_t)NTOK * INNER;
    h16* Vt  = Kb  + (size_t)NTOK * INNER;
    h16* Oh  = Vt  + (size_t)NTOK * INNER;
    h16* Wqp = Oh  + (size_t)NTOK * INNER;
    h16* Wkp = Wqp + (size_t)QD * INNER;
    h16* Wvp = Wkp + (size_t)QD * INNER;
    h16* Wop = Wvp + (size_t)QD * INNER;
    float* maskb = (float*)(Wop + (size_t)INNER * QD);

    pack_x_kernel<<<dim3((NTOK * QD + 255) / 256, 2), 256, 0, stream>>>(hs, ehs, Xh, Eh);
    pack_w_kernel<<<320, 128, 0, stream>>>(Wq, Wqp, QD, INNER);
    pack_w_kernel<<<320, 128, 0, stream>>>(Wk, Wkp, QD, INNER);
    pack_w_kernel<<<320, 128, 0, stream>>>(Wv, Wvp, QD, INNER);
    pack_w_kernel<<<320, 128, 0, stream>>>(Wo, Wop, INNER, QD);
    mask_kernel<<<16, 256, 0, stream>>>(lam, maskb);

    gemm_qkv_kernel<<<dim3(256, 2, 3), 128, 0, stream>>>(Xh, Eh, Wqp, Wkp, Wvp, Qb, Kb, Vt);
    attn_kernel<<<512, 128, 0, stream>>>(Qb, Kb, Vt, maskb, Oh);
    gemm_out_kernel<<<dim3(256, 5), 128, 0, stream>>>(Oh, Wop, bo, out);
}
